// SelfMultiHeadAttention_28312424415312
// MI455X (gfx1250) — matrix operand order repaired and checked by reading, not compiled, not run on hardware
//
#include <hip/hip_runtime.h>

// ---------------------------------------------------------------------------
// Types / helpers
// ---------------------------------------------------------------------------
typedef __attribute__((ext_vector_type(16))) __bf16 v16bf;
typedef __attribute__((ext_vector_type(8)))  float  v8f;

union Frag {
    v16bf v;
    uint4 q[2];
};

__device__ __forceinline__ unsigned short f32_to_bf16(float f) {
    union { float f; unsigned u; } c; c.f = f;
    unsigned r = c.u + 0x7FFFu + ((c.u >> 16) & 1u);   // round to nearest even
    return (unsigned short)(r >> 16);
}

// Low 32 bits of a generic pointer into LDS == LDS byte offset (flat-LDS
// aperture truncates to addr[31:0]).
__device__ __forceinline__ unsigned lds_off(const void* p) {
    return (unsigned)(reinterpret_cast<uintptr_t>(p));
}

// CDNA5 async global->LDS copy, 16 bytes per lane. Tracked by ASYNCcnt.
__device__ __forceinline__ void async_copy_b128(unsigned lds_addr, const void* gaddr) {
    asm volatile("global_load_async_to_lds_b128 %0, %1, off"
                 :: "v"(lds_addr), "v"(gaddr) : "memory");
}
__device__ __forceinline__ void wait_async0() {
    asm volatile("s_wait_asynccnt 0x0" ::: "memory");
}

// A-matrix fragment (16x32 bf16) from a row-major matrix (global or LDS).
// ISA 7.12.2: lanes 0-15 -> row = lane, K runs {0..7, 16..23};
// lanes 16-31 -> row = lane-16, K runs {8..15, 24..31}.
__device__ __forceinline__ Frag load_fragA_rowmajor(const unsigned short* base, int ld,
                                                    int row0, int k0, int lane) {
    int r = lane & 15, half = lane >> 4;
    const unsigned short* p = base + (size_t)(row0 + r) * ld + k0 + (half ? 8 : 0);
    Frag f;
    f.q[0] = *reinterpret_cast<const uint4*>(p);
    f.q[1] = *reinterpret_cast<const uint4*>(p + 16);
    return f;
}

// B-matrix fragment (32x16 bf16). Lanes 0-15: col = lane, K runs {0..7, 16..23};
// lanes 16-31: col = lane-16, K runs {8..15, 24..31}. Source must be K-contiguous per column
// with column stride `ld`.
__device__ __forceinline__ Frag load_fragB_kcontig(const unsigned short* base, int ld,
                                                   int col0, int k0, int lane) {
    int r = lane & 15, half = lane >> 4;
    const unsigned short* p = base + (size_t)(col0 + r) * ld + k0 + (half ? 8 : 0);
    Frag f;
    f.q[0] = *reinterpret_cast<const uint4*>(p);
    f.q[1] = *reinterpret_cast<const uint4*>(p + 16);
    return f;
}

__device__ __forceinline__ v8f wmma_bf16(const Frag& a, const Frag& b, v8f c) {
    return __builtin_amdgcn_wmma_f32_16x16x32_bf16(false, a.v, false, b.v,
                                                   (short)0, c, false, false);
}

// ---------------------------------------------------------------------------
// Conversion kernels
// ---------------------------------------------------------------------------
// W f32 [K,N] row-major  ->  WT bf16 [N,K] (K-contiguous per output column).
__global__ void cvt_w_transpose_kernel(const float* __restrict__ src,
                                       unsigned short* __restrict__ dst,
                                       int K, int N) {
    int i = blockIdx.x * 256 + threadIdx.x;
    if (i >= K * N) return;
    int k = i / N, n = i - k * N;
    dst[(size_t)n * K + k] = f32_to_bf16(src[i]);
}

// x (f32 [8192,512]) -> xb (bf16 [8192,512]) and first half of concat buffer
// hin (bf16 [8192,1024]).
__global__ void pack_x_kernel(const float* __restrict__ x,
                              unsigned short* __restrict__ xb,
                              unsigned short* __restrict__ hin) {
    int i = blockIdx.x * 256 + threadIdx.x;
    int row = i >> 9, col = i & 511;
    unsigned short v = f32_to_bf16(x[i]);
    xb[i] = v;
    hin[(size_t)row * 1024 + col] = v;
}

// ---------------------------------------------------------------------------
// bf16 WMMA GEMM: C[M,N] = A[M,K] @ BT[N,K]^T (+bias) (+residual)
// A row-major, BT transposed (K-contiguous per column).
// Block: 256 threads (8 waves), tile 128x64, wave = 32x32 (2x2 frags).
// A tile is double-buffered in LDS and filled with async global->LDS copies;
// B fragments come straight from global (weights are L2-resident).
// ---------------------------------------------------------------------------
__global__ __launch_bounds__(256) void gemm_bf16_kernel(
    const unsigned short* __restrict__ A, int lda,
    const unsigned short* __restrict__ BT, int ldb,   // ldb == K
    float* __restrict__ Cf, unsigned short* __restrict__ Cb, int ldc,
    const float* __restrict__ bias,
    const float* __restrict__ residual, int ldres,
    int K) {
    __shared__ __align__(16) unsigned short As[2][128 * 32];  // row-major [128][32]

    int tid = threadIdx.x, lane = tid & 31, wid = tid >> 5;
    int wm = wid >> 1, wn = wid & 1;
    int row0 = blockIdx.y * 128, col0 = blockIdx.x * 64;

    // Each thread stages 32 bytes of the A tile per K-step via 2 async b128s.
    int arow = tid >> 1, ac = (tid & 1) * 16;
    const unsigned short* agp = A + (size_t)(row0 + arow) * lda + ac;

    v8f acc[2][2];
    acc[0][0] = {}; acc[0][1] = {}; acc[1][0] = {}; acc[1][1] = {};

    // Prologue: kick off tile 0.
    {
        unsigned l = lds_off(&As[0][arow * 32 + ac]);
        async_copy_b128(l, agp);
        async_copy_b128(l + 16, agp + 8);
    }

    int KT = K >> 5;
    for (int kt = 0; kt < KT; ++kt) {
        int p = kt & 1;
        wait_async0();        // my async writes into As[p] are done
        __syncthreads();      // everyone's are done; everyone left As[1-p]

        if (kt + 1 < KT) {    // prefetch next tile into the other buffer
            const unsigned short* g = agp + (size_t)(kt + 1) * 32;
            unsigned l = lds_off(&As[1 - p][arow * 32 + ac]);
            async_copy_b128(l, g);
            async_copy_b128(l + 16, g + 8);
        }

        Frag a0 = load_fragA_rowmajor(As[p], 32, wm * 32 + 0, 0, lane);
        Frag a1 = load_fragA_rowmajor(As[p], 32, wm * 32 + 16, 0, lane);
        Frag b0 = load_fragB_kcontig(BT, ldb, col0 + wn * 32 + 0, kt * 32, lane);
        Frag b1 = load_fragB_kcontig(BT, ldb, col0 + wn * 32 + 16, kt * 32, lane);

        acc[0][0] = wmma_bf16(a0, b0, acc[0][0]);
        acc[0][1] = wmma_bf16(a0, b1, acc[0][1]);
        acc[1][0] = wmma_bf16(a1, b0, acc[1][0]);
        acc[1][1] = wmma_bf16(a1, b1, acc[1][1]);
    }

    // Epilogue. C-frag layout: VGPR v, lanes 0-15 -> (M=v, N=lane);
    // lanes 16-31 -> (M=v+8, N=lane-16).
    int r = lane & 15, half = lane >> 4;
#pragma unroll
    for (int im = 0; im < 2; ++im)
#pragma unroll
        for (int in = 0; in < 2; ++in)
#pragma unroll
            for (int v = 0; v < 8; ++v) {
                int row = row0 + wm * 32 + im * 16 + v + half * 8;
                int col = col0 + wn * 32 + in * 16 + r;
                float val = acc[im][in][v];
                if (bias) val += bias[col];
                if (residual) val += residual[(size_t)row * ldres + col];
                if (Cb) Cb[(size_t)row * ldc + col] = f32_to_bf16(val);
                else    Cf[(size_t)row * ldc + col] = val;
            }
}

// ---------------------------------------------------------------------------
// RoPE + split/scatter of the fused QKV result (bias already added in GEMM).
// qkv f32 [B*N, 1536] laid out [h(8)][hd(64)][which(3)]; enc f32 [2,B,1,N,64].
// Writes q,k bf16 [B,H,N,64]; v bf16 transposed [B,H,64,N].
// ---------------------------------------------------------------------------
__global__ void rope_kernel(const float* __restrict__ qkv,
                            const float* __restrict__ enc,
                            unsigned short* __restrict__ qo,
                            unsigned short* __restrict__ ko,
                            unsigned short* __restrict__ vT) {
    int idx = blockIdx.x * 256 + threadIdx.x;   // [b][h][n][hd]
    int hd = idx & 63;
    int n  = (idx >> 6) & 2047;
    int h  = (idx >> 17) & 7;
    int b  = idx >> 20;

    size_t row = (size_t)b * 2048 + n;
    int colq = h * 192 + hd * 3;
    float qv = qkv[row * 1536 + colq + 0];
    float kv = qkv[row * 1536 + colq + 1];
    float vv = qkv[row * 1536 + colq + 2];
    int colp = h * 192 + (hd ^ 1) * 3;
    float qp = qkv[row * 1536 + colp + 0];
    float kp = qkv[row * 1536 + colp + 1];
    float rq = (hd & 1) ? qp : -qp;      // rotate_half
    float rk = (hd & 1) ? kp : -kp;

    size_t eb = ((size_t)b * 2048 + n) * 64 + hd;
    float f0 = enc[eb];
    float f1 = enc[(size_t)4 * 2048 * 64 + eb];

    size_t qi = (((size_t)(b * 8 + h) * 2048) + n) * 64 + hd;
    qo[qi] = f32_to_bf16(qv * f0 + rq * f1);
    ko[qi] = f32_to_bf16(kv * f0 + rk * f1);
    vT[((size_t)(b * 8 + h) * 64 + hd) * 2048 + n] = f32_to_bf16(vv);
}

// ---------------------------------------------------------------------------
// Flash attention. Grid: B*H*(N/64) blocks, 128 threads (4 waves).
// Each wave owns 16 q rows; loops over 64-key KV tiles with online softmax.
// q,k: bf16 [BH, 2048, 64]; vT: bf16 [BH, 64, 2048]; o: bf16 [B*N, 512].
// ---------------------------------------------------------------------------
__global__ __launch_bounds__(128) void attn_kernel(
    const unsigned short* __restrict__ qg,
    const unsigned short* __restrict__ kg,
    const unsigned short* __restrict__ vTg,
    unsigned short* __restrict__ og) {
    __shared__ __align__(16) unsigned short P[4][16 * 64];  // per-wave P tile

    int lane = threadIdx.x & 31, w = threadIdx.x >> 5;
    int bh = blockIdx.x >> 5;            // 32 q-tiles of 64 rows each
    int qt = blockIdx.x & 31;
    int n0 = qt * 64 + w * 16;

    const unsigned short* qb = qg  + (size_t)bh * 2048 * 64;
    const unsigned short* kb = kg  + (size_t)bh * 2048 * 64;
    const unsigned short* vb = vTg + (size_t)bh * 64 * 2048;

    Frag qa0 = load_fragA_rowmajor(qb, 64, n0, 0, lane);
    Frag qa1 = load_fragA_rowmajor(qb, 64, n0, 32, lane);

    v8f acc[4];
    acc[0] = {}; acc[1] = {}; acc[2] = {}; acc[3] = {};
    float mrow[8], lrow[8];
#pragma unroll
    for (int v = 0; v < 8; ++v) { mrow[v] = -3.402823466e+38f; lrow[v] = 0.f; }

    int r = lane & 15, half = lane >> 4;

    for (int j = 0; j < 2048; j += 64) {
        // ---- S = Q K^T for a 16x64 tile (4 column sub-tiles) ----
        v8f s[4];
#pragma unroll
        for (int ct = 0; ct < 4; ++ct) {
            Frag b0 = load_fragB_kcontig(kb, 64, j + ct * 16, 0, lane);
            Frag b1 = load_fragB_kcontig(kb, 64, j + ct * 16, 32, lane);
            v8f z = {};
            z = wmma_bf16(qa0, b0, z);
            s[ct] = wmma_bf16(qa1, b1, z);
        }
        // ---- online softmax ----
        float mnew[8], sc[8];
#pragma unroll
        for (int v = 0; v < 8; ++v) {
            s[0][v] *= 0.125f; s[1][v] *= 0.125f; s[2][v] *= 0.125f; s[3][v] *= 0.125f;
            float x = fmaxf(fmaxf(s[0][v], s[1][v]), fmaxf(s[2][v], s[3][v]));
            x = fmaxf(x, __shfl_xor(x, 1, 32));
            x = fmaxf(x, __shfl_xor(x, 2, 32));
            x = fmaxf(x, __shfl_xor(x, 4, 32));
            x = fmaxf(x, __shfl_xor(x, 8, 32));   // reduce over 16-lane half
            mnew[v] = fmaxf(mrow[v], x);
        }
#pragma unroll
        for (int v = 0; v < 8; ++v) {
            s[0][v] = __expf(s[0][v] - mnew[v]);
            s[1][v] = __expf(s[1][v] - mnew[v]);
            s[2][v] = __expf(s[2][v] - mnew[v]);
            s[3][v] = __expf(s[3][v] - mnew[v]);
            float t = s[0][v] + s[1][v] + s[2][v] + s[3][v];
            t += __shfl_xor(t, 1, 32);
            t += __shfl_xor(t, 2, 32);
            t += __shfl_xor(t, 4, 32);
            t += __shfl_xor(t, 8, 32);
            sc[v] = __expf(mrow[v] - mnew[v]);
            lrow[v] = lrow[v] * sc[v] + t;
            mrow[v] = mnew[v];
        }
#pragma unroll
        for (int t = 0; t < 4; ++t)
#pragma unroll
            for (int v = 0; v < 8; ++v) acc[t][v] *= sc[v];

        // ---- P (C-layout) -> LDS -> A-layout fragments ----
#pragma unroll
        for (int ct = 0; ct < 4; ++ct)
#pragma unroll
            for (int v = 0; v < 8; ++v)
                P[w][(v + half * 8) * 64 + ct * 16 + r] = f32_to_bf16(s[ct][v]);

        Frag pa0 = load_fragA_rowmajor(P[w], 64, 0, 0, lane);
        Frag pa1 = load_fragA_rowmajor(P[w], 64, 0, 32, lane);

        // ---- O += P @ V ----
#pragma unroll
        for (int t = 0; t < 4; ++t) {
            Frag v0 = load_fragB_kcontig(vb, 2048, t * 16, j, lane);
            Frag v1 = load_fragB_kcontig(vb, 2048, t * 16, j + 32, lane);
            acc[t] = wmma_bf16(pa0, v0, acc[t]);
            acc[t] = wmma_bf16(pa1, v1, acc[t]);
        }
    }

    // ---- write O (normalized) to [B*N, 512] bf16 ----
    int b = bh >> 3, h = bh & 7;
#pragma unroll
    for (int t = 0; t < 4; ++t)
#pragma unroll
        for (int v = 0; v < 8; ++v) {
            int n = n0 + v + half * 8;
            int hd = t * 16 + r;
            og[((size_t)(b * 2048 + n)) * 512 + h * 64 + hd] =
                f32_to_bf16(acc[t][v] / lrow[v]);
        }
}

// ---------------------------------------------------------------------------
// LayerNorm (over 1024) + exact GELU, output bf16. One block per row.
// ---------------------------------------------------------------------------
__global__ __launch_bounds__(256) void ln_gelu_kernel(
    const float* __restrict__ hbuf,
    const float* __restrict__ gamma,
    const float* __restrict__ beta,
    unsigned short* __restrict__ out) {
    __shared__ float rsum[256];
    __shared__ float rsq[256];
    int tid = threadIdx.x;
    size_t base = (size_t)blockIdx.x * 1024;

    float4 x = reinterpret_cast<const float4*>(hbuf + base)[tid];
    float s = x.x + x.y + x.z + x.w;
    float q = x.x * x.x + x.y * x.y + x.z * x.z + x.w * x.w;
    rsum[tid] = s; rsq[tid] = q;
    __syncthreads();
    for (int st = 128; st > 0; st >>= 1) {
        if (tid < st) { rsum[tid] += rsum[tid + st]; rsq[tid] += rsq[tid + st]; }
        __syncthreads();
    }
    float mean = rsum[0] * (1.0f / 1024.0f);
    float var  = rsq[0] * (1.0f / 1024.0f) - mean * mean;
    float rstd = rsqrtf(var + 1e-5f);

    float e[4] = {x.x, x.y, x.z, x.w};
#pragma unroll
    for (int k = 0; k < 4; ++k) {
        int c = tid * 4 + k;
        float y = (e[k] - mean) * rstd * gamma[c] + beta[c];
        float g = 0.5f * y * (1.0f + erff(y * 0.70710678118654752f));
        out[base + c] = f32_to_bf16(g);
    }
}

// ---------------------------------------------------------------------------
// Host orchestration
// ---------------------------------------------------------------------------
extern "C" void kernel_launch(void* const* d_in, const int* in_sizes, int n_in,
                              void* d_out, int out_size, void* d_ws, size_t ws_size,
                              hipStream_t stream) {
    (void)in_sizes; (void)n_in; (void)out_size; (void)ws_size;

    const float* x[2]   = {(const float*)d_in[0], (const float*)d_in[1]};
    const float* enc[2] = {(const float*)d_in[2], (const float*)d_in[3]};
    const float* Wqkv  = (const float*)d_in[4];
    const float* bqkv  = (const float*)d_in[5];
    const float* Wproj = (const float*)d_in[6];
    const float* bproj = (const float*)d_in[7];
    const float* W1    = (const float*)d_in[8];
    const float* b1    = (const float*)d_in[9];
    const float* ln_g  = (const float*)d_in[10];
    const float* ln_b  = (const float*)d_in[11];
    const float* W2    = (const float*)d_in[12];
    const float* b2    = (const float*)d_in[13];
    float* out = (float*)d_out;

    char* ws = (char*)d_ws;
    const size_t MB = 1024 * 1024;
    // Transposed bf16 weights.
    unsigned short* wqT = (unsigned short*)(ws + 0);                 // [1536,512]
    unsigned short* wpT = (unsigned short*)(ws + 1572864);           // [512,512]
    unsigned short* w1T = (unsigned short*)(ws + 1572864 + 524288);  // [1024,1024]
    unsigned short* w2T = (unsigned short*)(ws + 1572864 + 524288 + 2097152); // [512,1024]
    unsigned short* xb   = (unsigned short*)(ws + 8 * MB);    // [8192,512] bf16
    unsigned short* qb   = (unsigned short*)(ws + 16 * MB);   // [B,H,N,64]
    unsigned short* kbuf = (unsigned short*)(ws + 24 * MB);
    unsigned short* vT   = (unsigned short*)(ws + 32 * MB);   // [B,H,64,N]
    unsigned short* ob   = (unsigned short*)(ws + 40 * MB);   // [B*N,512]
    unsigned short* hin  = (unsigned short*)(ws + 48 * MB);   // [8192,1024] bf16
    float*          qkvf = (float*)(ws + 64 * MB);            // [8192,1536] f32
    float*          hf   = (float*)(ws + 64 * MB);            // [8192,1024] f32 (reuse)
    unsigned short* gbuf = (unsigned short*)(ws + 96 * MB);   // [8192,1024] bf16

    // One-time weight conversion to transposed bf16.
    cvt_w_transpose_kernel<<<(512 * 1536) / 256, 256, 0, stream>>>(Wqkv, wqT, 512, 1536);
    cvt_w_transpose_kernel<<<(512 * 512) / 256, 256, 0, stream>>>(Wproj, wpT, 512, 512);
    cvt_w_transpose_kernel<<<(1024 * 1024) / 256, 256, 0, stream>>>(W1, w1T, 1024, 1024);
    cvt_w_transpose_kernel<<<(1024 * 512) / 256, 256, 0, stream>>>(W2, w2T, 1024, 512);

    for (int i = 0; i < 2; ++i) {
        pack_x_kernel<<<(8192 * 512) / 256, 256, 0, stream>>>(x[i], xb, hin);

        // qkv = x @ Wqkv + bqkv   (f32 out)
        gemm_bf16_kernel<<<dim3(1536 / 64, 8192 / 128), 256, 0, stream>>>(
            xb, 512, wqT, 512, qkvf, nullptr, 1536, bqkv, nullptr, 0, 512);

        rope_kernel<<<(4 * 8 * 2048 * 64) / 256, 256, 0, stream>>>(
            qkvf, enc[i], qb, kbuf, vT);

        attn_kernel<<<4 * 8 * (2048 / 64), 128, 0, stream>>>(qb, kbuf, vT, ob);

        // message = o @ Wproj + bproj  -> bf16 into second half of concat buf
        gemm_bf16_kernel<<<dim3(512 / 64, 8192 / 128), 256, 0, stream>>>(
            ob, 512, wpT, 512, nullptr, hin + 512, 1024, bproj, nullptr, 0, 512);

        // h = [x, message] @ W1 + b1  (f32 out)
        gemm_bf16_kernel<<<dim3(1024 / 64, 8192 / 128), 256, 0, stream>>>(
            hin, 1024, w1T, 1024, hf, nullptr, 1024, b1, nullptr, 0, 1024);

        ln_gelu_kernel<<<8192, 256, 0, stream>>>(hf, ln_g, ln_b, gbuf);

        // out = x + gelu(ln(h)) @ W2 + b2  (f32 out with fused residual)
        gemm_bf16_kernel<<<dim3(512 / 64, 8192 / 128), 256, 0, stream>>>(
            gbuf, 1024, w2T, 1024, out + (size_t)i * 8192 * 512, nullptr, 512,
            b2, x[i], 512, 1024);
    }
}
